// MambaMixer_76699525972316
// MI455X (gfx1250) — matrix operand order repaired and checked by reading, not compiled, not run on hardware
//
#include <hip/hip_runtime.h>
#include <hip/hip_bf16.h>

// ---------- dims ----------
#define LSEQ   2048
#define DMODEL 1024
#define DINNER 2048
#define DSTATE 16
#define DTRANK 64
#define DCONV  4
#define NGROUP 2
#define DBCW   (DTRANK + 2 * DSTATE)   // 96

typedef unsigned short ush;

// ---------- WMMA types ----------
typedef __attribute__((ext_vector_type(16))) __bf16 v16bf;
typedef __attribute__((ext_vector_type(8)))  float  v8f;

union FragAB { v16bf v; ush h[16]; uint4 q[2]; };

__device__ __forceinline__ ush f2bf(float f) {
    union { float f; unsigned u; } x; x.f = f;
    unsigned u = x.u;
    unsigned r = u + 0x7FFFu + ((u >> 16) & 1u);  // round-to-nearest-even
    return (ush)(r >> 16);
}

__device__ __forceinline__ float silu(float x) { return x / (1.f + expf(-x)); }

__device__ __forceinline__ v8f wmma_bf16(v16bf a, v16bf b, v8f c) {
    return __builtin_amdgcn_wmma_f32_16x16x32_bf16(
        false, a, false, b, (short)0, c, false, false);
}

// ---- A fragment from row-major bf16 (16x32 tile). Per ISA 7.12.2 the lane's
// 16 elements are two contiguous K-runs: [kh, kh+8) and [16+kh, 16+kh+8).
__device__ __forceinline__ v16bf load_a_bf(const ush* A, int row0, int lda,
                                           int kbase, int lane) {
    FragAB f;
    int m  = lane & 15;
    int kh = (lane >> 4) * 8;
    const ush* p = A + (size_t)(row0 + m) * lda + kbase + kh;
    f.q[0] = *reinterpret_cast<const uint4*>(p);
    f.q[1] = *reinterpret_cast<const uint4*>(p + 16);
    return f.v;
}

// ---- B fragment from pre-packed weights: lane's 16 bf16 are contiguous.
// Packed: P[((kt*NT + nt)*32 + lane)*16 + e] = B[kt*32 + kb(lane) + e + (e >= 8 ? 8 : 0), nt*16 + n(lane)]
__device__ __forceinline__ v16bf load_b_pk(const ush* P, int kt, int nt, int NT,
                                           int lane) {
    FragAB f;
    const ush* p = P + ((size_t)((size_t)kt * NT + nt) * 32 + lane) * 16;
    f.q[0] = *reinterpret_cast<const uint4*>(p);
    f.q[1] = *reinterpret_cast<const uint4*>(p + 8);
    return f.v;
}

__device__ __forceinline__ void store_d_f32(float* D, int row0, int col0, int ldd,
                                            int lane, const v8f acc) {
    int n = lane & 15, mh = (lane >> 4) * 8;
#pragma unroll
    for (int r = 0; r < 8; ++r)
        D[(size_t)(row0 + mh + r) * ldd + col0 + n] = acc[r];
}

__device__ __forceinline__ void store_d_bf(ush* D, int row0, int col0, int ldd,
                                           int lane, const v8f acc) {
    int n = lane & 15, mh = (lane >> 4) * 8;
#pragma unroll
    for (int r = 0; r < 8; ++r)
        D[(size_t)(row0 + mh + r) * ldd + col0 + n] = f2bf(acc[r]);
}

// ---- register-blocked 2x4-tile (32x64) GEMM block:
// 8 WMMAs per 12 b128 loads, ~112 live VGPRs -> no spills.
#define MI 2
#define NJ 4
__device__ __forceinline__ void gemm_block24(const ush* __restrict__ A,
                                             const ush* __restrict__ P,
                                             int m0, int nt0, int NT, int K,
                                             int lda, int lane, v8f acc[MI * NJ]) {
#pragma unroll
    for (int t = 0; t < MI * NJ; ++t) acc[t] = (v8f){};
    int KT = K >> 5;
    for (int kt = 0; kt < KT; ++kt) {
        if (kt + 1 < KT) {
            __builtin_prefetch(P + ((size_t)((size_t)(kt + 1) * NT + nt0) * 32 + lane) * 16, 0, 3);
            __builtin_prefetch(A + (size_t)(m0 + (lane & 15)) * lda + (kt + 1) * 32, 0, 3);
        }
        v16bf a[MI], b[NJ];
#pragma unroll
        for (int i = 0; i < MI; ++i) a[i] = load_a_bf(A, m0 + 16 * i, lda, kt * 32, lane);
#pragma unroll
        for (int j = 0; j < NJ; ++j) b[j] = load_b_pk(P, kt, nt0 + j, NT, lane);
#pragma unroll
        for (int i = 0; i < MI; ++i)
#pragma unroll
            for (int j = 0; j < NJ; ++j)
                acc[i * NJ + j] = wmma_bf16(a[i], b[j], acc[i * NJ + j]);
    }
}

// ================= one-time conversion / packing kernels =================

__global__ void k_cvt_bf16(const float* __restrict__ src, ush* __restrict__ dst,
                           int nElem) {
    int i = (blockIdx.x * blockDim.x + threadIdx.x) * 4;
    if (i + 3 < nElem) {
#pragma unroll
        for (int j = 0; j < 4; ++j) dst[i + j] = f2bf(src[i + j]);
    }
}

// pack row-major f32 B(K x N) into WMMA B-fragment order (bf16)
__global__ void k_pack_b(const float* __restrict__ B, ush* __restrict__ P,
                         int K, int N) {
    int idx  = blockIdx.x * blockDim.x + threadIdx.x;   // (K/32)*(N/16)*32 total
    int lane = idx & 31;
    int tile = idx >> 5;
    int NT = N >> 4;
    int kt = tile / NT, nt = tile % NT;
    int n  = nt * 16 + (lane & 15);
    int kb = kt * 32 + (lane >> 4) * 8;
    ush* dst = P + (size_t)idx * 16;
#pragma unroll
    for (int e = 0; e < 16; ++e)
        dst[e] = f2bf(B[(size_t)(kb + e + (e >= 8 ? 8 : 0)) * N + n]);
}

// ================= pipeline kernels =================

// 1) xz = x @ in_proj_w ; cols [0,2048) -> xi (bf16), [2048,4096) -> res (f32)
__global__ void k_in_proj(const ush* __restrict__ xb, const ush* __restrict__ wp,
                          ush* __restrict__ xi, float* __restrict__ res) {
    int lane = threadIdx.x & 31;
    int wid  = (blockIdx.x * blockDim.x + threadIdx.x) >> 5;
    const int N  = 2 * DINNER;
    const int NB = N / (16 * NJ);              // 64 blocks of 4 n-tiles
    int m0  = (wid / NB) * (16 * MI);
    int nt0 = (wid % NB) * NJ;
    v8f acc[MI * NJ];
    gemm_block24(xb, wp, m0, nt0, N / 16, DMODEL, DMODEL, lane, acc);
#pragma unroll
    for (int i = 0; i < MI; ++i)
#pragma unroll
        for (int j = 0; j < NJ; ++j) {
            int col = (nt0 + j) * 16;
            if (col < DINNER) store_d_bf (xi,  m0 + 16 * i, col,          DINNER, lane, acc[i * NJ + j]);
            else              store_d_f32(res, m0 + 16 * i, col - DINNER, DINNER, lane, acc[i * NJ + j]);
        }
}

// 2) grouped causal conv as im2col GEMM (K_flat = tap*1024 + ci), bias + SiLU
__device__ __forceinline__ v16bf load_a_conv_bf(const ush* xi, int l0, int g,
                                                int kb, int lane) {
    FragAB f;
    int m  = lane & 15;
    int kh = (lane >> 4) * 8;
    int t  = kb >> 10;                 // 32-chunk never crosses a tap
    int ib = kb & 1023;
    int lp = l0 + m - (DCONV - 1) + t;
    if (lp >= 0) {
        const ush* p = xi + (size_t)lp * DINNER + g * (DINNER / NGROUP) + ib + kh;
        f.q[0] = *reinterpret_cast<const uint4*>(p);
        f.q[1] = *reinterpret_cast<const uint4*>(p + 16);
    } else {
        f.q[0] = make_uint4(0u, 0u, 0u, 0u);
        f.q[1] = make_uint4(0u, 0u, 0u, 0u);
    }
    return f.v;
}

__global__ void k_conv(const ush* __restrict__ xi, const ush* __restrict__ wp,
                       const float* __restrict__ bias,
                       ush* __restrict__ xcb, float* __restrict__ xcf) {
    int lane = threadIdx.x & 31;
    int wid  = (blockIdx.x * blockDim.x + threadIdx.x) >> 5;
    const int NT = DINNER / 16;                // 128
    const int NB = DINNER / (16 * NJ);         // 32
    int m0  = (wid / NB) * (16 * MI);
    int nt0 = (wid % NB) * NJ;
    int g   = (nt0 * 16) >> 10;                // uniform: 1024 % 64 == 0
    const int KF = DCONV * (DINNER / NGROUP);  // 4096
    v8f acc[MI * NJ];
#pragma unroll
    for (int t = 0; t < MI * NJ; ++t) acc[t] = (v8f){};
    for (int kt = 0; kt < (KF >> 5); ++kt) {
        if (kt + 1 < (KF >> 5))
            __builtin_prefetch(wp + ((size_t)((size_t)(kt + 1) * NT + nt0) * 32 + lane) * 16, 0, 3);
        v16bf a[MI], b[NJ];
#pragma unroll
        for (int i = 0; i < MI; ++i) a[i] = load_a_conv_bf(xi, m0 + 16 * i, g, kt * 32, lane);
#pragma unroll
        for (int j = 0; j < NJ; ++j) b[j] = load_b_pk(wp, kt, nt0 + j, NT, lane);
#pragma unroll
        for (int i = 0; i < MI; ++i)
#pragma unroll
            for (int j = 0; j < NJ; ++j)
                acc[i * NJ + j] = wmma_bf16(a[i], b[j], acc[i * NJ + j]);
    }
    int n = lane & 15, mh = (lane >> 4) * 8;
#pragma unroll
    for (int i = 0; i < MI; ++i)
#pragma unroll
        for (int j = 0; j < NJ; ++j) {
            int col = (nt0 + j) * 16 + n;
            float bv = bias[col];
#pragma unroll
            for (int r = 0; r < 8; ++r) {
                float v = silu(acc[i * NJ + j][r] + bv);
                size_t o = (size_t)(m0 + 16 * i + mh + r) * DINNER + col;
                xcf[o] = v;
                xcb[o] = f2bf(v);
            }
        }
}

// 3) dbc = xc @ x_proj_w (N=96, small op: 1 tile per wave)
__global__ void k_xproj(const ush* __restrict__ xcb, const ush* __restrict__ wp,
                        float* __restrict__ dbc, ush* __restrict__ drb) {
    int lane = threadIdx.x & 31;
    int wid  = (blockIdx.x * blockDim.x + threadIdx.x) >> 5;
    const int NT = DBCW / 16;                  // 6
    int mt = wid / NT, nt = wid % NT;
    v8f acc = {};
    for (int kt = 0; kt < (DINNER >> 5); ++kt) {
        if (kt + 1 < (DINNER >> 5)) {
            __builtin_prefetch(wp + ((size_t)((size_t)(kt + 1) * NT + nt) * 32 + lane) * 16, 0, 3);
            __builtin_prefetch(xcb + (size_t)(mt * 16 + (lane & 15)) * DINNER + (kt + 1) * 32, 0, 3);
        }
        v16bf a = load_a_bf(xcb, mt * 16, DINNER, kt * 32, lane);
        v16bf b = load_b_pk(wp, kt, nt, NT, lane);
        acc = wmma_bf16(a, b, acc);
    }
    store_d_f32(dbc, mt * 16, nt * 16, DBCW, lane, acc);
    if (nt < DTRANK / 16)                      // uniform per wave
        store_d_bf(drb, mt * 16, nt * 16, DTRANK, lane, acc);
}

// 4) delta = softplus(delta_r @ dt_proj_w + b), 2x4 blocked (K=64)
__global__ void k_dtproj(const ush* __restrict__ drb, const ush* __restrict__ wp,
                         const float* __restrict__ b, float* __restrict__ delta) {
    int lane = threadIdx.x & 31;
    int wid  = (blockIdx.x * blockDim.x + threadIdx.x) >> 5;
    const int NB = DINNER / (16 * NJ);         // 32
    int m0  = (wid / NB) * (16 * MI);
    int nt0 = (wid % NB) * NJ;
    v8f acc[MI * NJ];
    gemm_block24(drb, wp, m0, nt0, DINNER / 16, DTRANK, DTRANK, lane, acc);
    int n = lane & 15, mh = (lane >> 4) * 8;
#pragma unroll
    for (int i = 0; i < MI; ++i)
#pragma unroll
        for (int j = 0; j < NJ; ++j) {
            int col = (nt0 + j) * 16 + n;
            float bv = b[col];
#pragma unroll
            for (int r = 0; r < 8; ++r) {
                float v = acc[i * NJ + j][r] + bv;
                float sp = (v > 20.f) ? v : log1pf(expf(v));
                delta[(size_t)(m0 + 16 * i + mh + r) * DINNER + col] = sp;
            }
        }
}

// 5) selective scan: one channel per lane, h[16] in registers; output bf16
__global__ void k_scan(const float* __restrict__ delta, const float* __restrict__ xcf,
                       const float* __restrict__ dbc, const float* __restrict__ res,
                       const float* __restrict__ A_log, const float* __restrict__ Dp,
                       ush* __restrict__ yb) {
    int d = blockIdx.x * blockDim.x + threadIdx.x;   // 0..2047
    float Ad[DSTATE];
#pragma unroll
    for (int n = 0; n < DSTATE; ++n) Ad[n] = -expf(A_log[(size_t)d * DSTATE + n]);
    float Dv = Dp[d];
    float h[DSTATE];
#pragma unroll
    for (int n = 0; n < DSTATE; ++n) h[n] = 0.f;

    for (int l = 0; l < LSEQ; ++l) {
        float dl = delta[(size_t)l * DINNER + d];
        float u  = xcf[(size_t)l * DINNER + d];
        float du = dl * u;
        const float* bc = dbc + (size_t)l * DBCW;
        float y = 0.f;
#pragma unroll
        for (int n = 0; n < DSTATE; ++n) {
            float hn = expf(dl * Ad[n]) * h[n] + du * bc[DTRANK + n];
            h[n] = hn;
            y += hn * bc[DTRANK + DSTATE + n];
        }
        y += u * Dv;
        float r = res[(size_t)l * DINNER + d];
        yb[(size_t)l * DINNER + d] = f2bf(y * silu(r));
    }
}

// 6) out = yb @ out_proj_w  (f32 output), 2x4 blocked
__global__ void k_outproj(const ush* __restrict__ yb, const ush* __restrict__ wp,
                          float* __restrict__ out) {
    int lane = threadIdx.x & 31;
    int wid  = (blockIdx.x * blockDim.x + threadIdx.x) >> 5;
    const int NB = DMODEL / (16 * NJ);         // 16
    int m0  = (wid / NB) * (16 * MI);
    int nt0 = (wid % NB) * NJ;
    v8f acc[MI * NJ];
    gemm_block24(yb, wp, m0, nt0, DMODEL / 16, DINNER, DINNER, lane, acc);
#pragma unroll
    for (int i = 0; i < MI; ++i)
#pragma unroll
        for (int j = 0; j < NJ; ++j)
            store_d_f32(out, m0 + 16 * i, (nt0 + j) * 16, DMODEL, lane, acc[i * NJ + j]);
}

// ================= host launcher =================
extern "C" void kernel_launch(void* const* d_in, const int* in_sizes, int n_in,
                              void* d_out, int out_size, void* d_ws, size_t ws_size,
                              hipStream_t stream) {
    const float* x         = (const float*)d_in[0];
    const float* in_proj_w = (const float*)d_in[1];
    const float* conv_w    = (const float*)d_in[2];
    const float* conv_b    = (const float*)d_in[3];
    const float* x_proj_w  = (const float*)d_in[4];
    const float* dt_proj_w = (const float*)d_in[5];
    const float* dt_proj_b = (const float*)d_in[6];
    const float* A_log     = (const float*)d_in[7];
    const float* Dp        = (const float*)d_in[8];
    const float* out_proj_w= (const float*)d_in[9];
    float* out = (float*)d_out;

    char* ws = (char*)d_ws;
    size_t off = 0;
    auto take = [&](size_t bytes) { char* p = ws + off; off += (bytes + 255) & ~(size_t)255; return p; };

    ush*   xb    = (ush*)  take((size_t)LSEQ * DMODEL * 2);          // x bf16
    ush*   wb1   = (ush*)  take((size_t)DMODEL * 2 * DINNER * 2);    // packed in_proj_w
    ush*   wcb   = (ush*)  take((size_t)(DCONV * DINNER / NGROUP) * DINNER * 2); // packed conv_w
    ush*   wxb   = (ush*)  take((size_t)DINNER * DBCW * 2);          // packed x_proj_w
    ush*   wdtb  = (ush*)  take((size_t)DTRANK * DINNER * 2);        // packed dt_proj_w
    ush*   wob   = (ush*)  take((size_t)DINNER * DMODEL * 2);        // packed out_proj_w
    ush*   xi    = (ush*)  take((size_t)LSEQ * DINNER * 2);          // bf16 (reused as yb)
    float* res   = (float*)take((size_t)LSEQ * DINNER * 4);
    ush*   xcb   = (ush*)  take((size_t)LSEQ * DINNER * 2);
    float* xcf   = (float*)take((size_t)LSEQ * DINNER * 4);
    float* dbc   = (float*)take((size_t)LSEQ * DBCW * 4);
    ush*   drb   = (ush*)  take((size_t)LSEQ * DTRANK * 2);
    float* delta = (float*)take((size_t)LSEQ * DINNER * 4);
    ush*   yb    = xi;  // xi dead after k_conv

    dim3 blk(256);

    // --- one-time conversions / weight packing ---
    k_cvt_bf16<<<(LSEQ * DMODEL) / (256 * 4), blk, 0, stream>>>(x, xb, LSEQ * DMODEL);
    k_pack_b<<<((DMODEL / 32) * ((2 * DINNER) / 16) * 32) / 256, blk, 0, stream>>>(in_proj_w, wb1, DMODEL, 2 * DINNER);
    k_pack_b<<<(((DCONV * DINNER / NGROUP) / 32) * (DINNER / 16) * 32) / 256, blk, 0, stream>>>(conv_w, wcb, DCONV * DINNER / NGROUP, DINNER);
    k_pack_b<<<((DINNER / 32) * (DBCW / 16) * 32) / 256, blk, 0, stream>>>(x_proj_w, wxb, DINNER, DBCW);
    k_pack_b<<<((DTRANK / 32) * (DINNER / 16) * 32) / 256, blk, 0, stream>>>(dt_proj_w, wdtb, DTRANK, DINNER);
    k_pack_b<<<((DINNER / 32) * (DMODEL / 16) * 32) / 256, blk, 0, stream>>>(out_proj_w, wob, DINNER, DMODEL);

    // --- pipeline (stream-ordered) ---
    k_in_proj<<<((LSEQ / 32) * ((2 * DINNER) / 64)) / 8, blk, 0, stream>>>(xb, wb1, xi, res);
    k_conv<<<((LSEQ / 32) * (DINNER / 64)) / 8, blk, 0, stream>>>(xi, wcb, conv_b, xcb, xcf);
    k_xproj<<<((LSEQ / 16) * (DBCW / 16)) / 8, blk, 0, stream>>>(xcb, wxb, dbc, drb);
    k_dtproj<<<((LSEQ / 32) * (DINNER / 64)) / 8, blk, 0, stream>>>(drb, wdtb, dt_proj_b, delta);
    k_scan<<<DINNER / 256, blk, 0, stream>>>(delta, xcf, dbc, res, A_log, Dp, yb);
    k_outproj<<<((LSEQ / 32) * (DMODEL / 64)) / 8, blk, 0, stream>>>(yb, wob, out);
}
